// EMOJI_ATTENTION_LSTM_25555055411492
// MI455X (gfx1250) — matrix operand order repaired and checked by reading, not compiled, not run on hardware
//
#include <hip/hip_runtime.h>
#include <hip/hip_bf16.h>
#include <math.h>

// ---------------------------------------------------------------------------
// Sizes (fixed by the reference)
// ---------------------------------------------------------------------------
#define EMB   1024
#define HID   1024
#define SEQ   2048
#define NEMO  16
#define G4    (4*HID)      // 4096
#define LSTM_WGS 32        // persistent workgroups for the recurrence

typedef __attribute__((ext_vector_type(16))) __bf16 v16bf;
typedef __attribute__((ext_vector_type(8)))  float  v8f;

union FragBF { unsigned int u[8]; v16bf v; };

__device__ __forceinline__ unsigned short f2bf(float x) {
  unsigned u = __float_as_uint(x);
  unsigned r = (u + 0x7fffu + ((u >> 16) & 1u)) >> 16;
  return (unsigned short)r;
}
__device__ __forceinline__ float bf2f(unsigned v) {   // low 16 bits = bf16
  return __uint_as_float((v & 0xffffu) << 16);
}
__device__ __forceinline__ float bf2f_hi(unsigned v) { // high 16 bits = bf16
  return __uint_as_float(v & 0xffff0000u);
}
__device__ __forceinline__ float sigm(float x) { return 1.f / (1.f + expf(-x)); }

// gfx1250 async global->LDS copy (16B), tracked by ASYNCcnt (ISA 15.18.3).
// ldsoff: byte offset within the wave's LDS allocation (low 32 bits of a
// generic shared pointer); gaddr: 64-bit global address.
__device__ __forceinline__ void async_ld_b128(unsigned ldsoff, const void* gaddr) {
  asm volatile("global_load_async_to_lds_b128 %0, %1, off"
               :: "v"(ldsoff), "v"(gaddr) : "memory");
}
__device__ __forceinline__ void wait_async0() {
  asm volatile("s_wait_asynccnt 0x0" ::: "memory");
}
__device__ __forceinline__ unsigned lds_off(const void* p) {
  return (unsigned)(size_t)p;   // shared aperture: addr[31:0] == LDS offset
}

// ---------------------------------------------------------------------------
// K0: emoji mean, zero h buffers, bias_sum = b_ih + b_hh, zero step counters.
// Attention weights are mathematically unused: softmax of a single logit == 1,
// so attn_applied == emoji_ave exactly.
// ---------------------------------------------------------------------------
__global__ __launch_bounds__(256) void k_prep_small(
    const int* __restrict__ emoji_ids, const float* __restrict__ emoji_emb,
    const float* __restrict__ b_ih, const float* __restrict__ b_hh,
    float* __restrict__ emoji_ave, float* __restrict__ hbuf,
    float* __restrict__ bias_sum, unsigned* __restrict__ stepcnt)
{
  int tid = threadIdx.x;
  for (int e = tid; e < EMB; e += 256) {
    float s = 0.f;
    #pragma unroll
    for (int j = 0; j < NEMO; ++j)
      s += emoji_emb[(size_t)emoji_ids[j] * EMB + e];
    emoji_ave[e] = s * (1.f / NEMO);
  }
  for (int e = tid; e < 2 * HID; e += 256) hbuf[e] = 0.f;
  for (int e = tid; e < G4; e += 256) bias_sum[e] = b_ih[e] + b_hh[e];
  for (int e = tid; e < SEQ; e += 256) stepcnt[e] = 0u;
}

// ---------------------------------------------------------------------------
// K1: const_vec[i] = comb_W[i, EMB:] @ emoji_ave + comb_b[i]   (1024 rows)
// ---------------------------------------------------------------------------
__global__ __launch_bounds__(256) void k_constvec(
    const float* __restrict__ comb_W, const float* __restrict__ comb_b,
    const float* __restrict__ emoji_ave, float* __restrict__ const_vec)
{
  int i = blockIdx.x * 256 + threadIdx.x;      // 4 blocks * 256 = 1024 rows
  const float* row = comb_W + (size_t)i * (2 * EMB) + EMB;
  float s = comb_b[i];
  #pragma unroll 8
  for (int k = 0; k < EMB; ++k) s += row[k] * emoji_ave[k];
  const_vec[i] = s;
}

// ---------------------------------------------------------------------------
// K2: gather sentence embeddings to bf16, convert comb_W[:, :EMB] and W_ih
//     to bf16 (grid-stride over 7,340,032 elements).
// ---------------------------------------------------------------------------
__global__ __launch_bounds__(256) void k_convert_gather(
    const int* __restrict__ sent_ids, const float* __restrict__ word_emb,
    const float* __restrict__ comb_W, const float* __restrict__ W_ih,
    unsigned short* __restrict__ Xbf, unsigned short* __restrict__ Wcxbf,
    unsigned short* __restrict__ Wihbf)
{
  const int NX = SEQ * EMB;            // 2,097,152
  const int NC = EMB * EMB;            // 1,048,576
  const int NW = G4 * EMB;             // 4,194,304
  for (int i = blockIdx.x * 256 + threadIdx.x; i < NX + NC + NW;
       i += gridDim.x * 256) {
    if (i < NX) {
      int t = i >> 10, k = i & 1023;
      Xbf[i] = f2bf(word_emb[(size_t)sent_ids[t] * EMB + k]);
    } else if (i < NX + NC) {
      int j = i - NX;
      int r = j >> 10, k = j & 1023;
      Wcxbf[j] = f2bf(comb_W[(size_t)r * (2 * EMB) + k]);
    } else {
      int j = i - NX - NC;
      Wihbf[j] = f2bf(W_ih[j]);
    }
  }
}

// ---------------------------------------------------------------------------
// K3/K4: WMMA "NT" GEMM:  C[i][t] = sum_k A[i][k] * B[t][k]
//   A: [M][K] bf16 row-major, B: [N][K] bf16 row-major, K multiple of 64.
//   128x128 tile per WG, 8 waves each 64x32 (4x2 WMMA 16x16x32 frags).
//   Double-buffered LDS pipeline, 64-deep K stage, staged with async
//   global->LDS b128 loads (no VGPR staging, no spills); one
//   s_wait_asynccnt + barrier per stage overlaps loads with 16 WMMAs.
//   MODE 0: out_bf[t][i] = bf16(relu(C + vec[i]))     (combined, for GEMM2)
//   MODE 1: out_f [t][i] = C + vec[i]                 (gates G, fp32)
// ---------------------------------------------------------------------------
template<int MODE>
__global__ __launch_bounds__(256) void k_gemm_nt(
    const unsigned short* __restrict__ A, const unsigned short* __restrict__ B,
    const float* __restrict__ vec, unsigned short* __restrict__ outBf,
    float* __restrict__ outF, int M, int N, int K)
{
  // 2 stages x 128 rows x 64 bf16 per matrix: 2*16KB*2 = 64KB LDS.
  __shared__ __align__(16) unsigned short As[2][128 * 64];
  __shared__ __align__(16) unsigned short Bs[2][128 * 64];

  const int tid  = threadIdx.x;
  const int lane = tid & 31;
  const int wave = tid >> 5;
  const int tiles_n = N >> 7;
  const int m0 = (blockIdx.x / tiles_n) * 128;
  const int n0 = (blockIdx.x % tiles_n) * 128;
  const int wm = (wave >> 2) * 64;     // wave M offset: 0 / 64
  const int wn = (wave & 3) * 32;      // wave N offset: 0 / 32 / 64 / 96

  v8f acc[4][2];
  #pragma unroll
  for (int i = 0; i < 4; ++i)
    #pragma unroll
    for (int j = 0; j < 2; ++j) acc[i][j] = {};

  const int mrow = lane & 15;
  const int kbA  = (lane < 16) ? 0 : 8;    // bf16 A: lane-half K interleave 0/8
  const int kbB  = (lane < 16) ? 0 : 8;    // bf16 B: lane-half K interleave 0/8

  // Staging geometry: 1024 16B-chunks per matrix per stage, 4 per thread.
  int srow[4], scol[4];
  #pragma unroll
  for (int s = 0; s < 4; ++s) {
    int c = tid + s * 256;
    srow[s] = c >> 3;
    scol[s] = (c & 7) << 3;
  }

  // Issue one stage's async global->LDS copies into buffer `buf` at K offset k0.
  auto stage_async = [&](int buf, int k0) {
    #pragma unroll
    for (int s = 0; s < 4; ++s) {
      async_ld_b128(lds_off(&As[buf][srow[s] * 64 + scol[s]]),
                    &A[(size_t)(m0 + srow[s]) * K + k0 + scol[s]]);
      async_ld_b128(lds_off(&Bs[buf][srow[s] * 64 + scol[s]]),
                    &B[(size_t)(n0 + srow[s]) * K + k0 + scol[s]]);
    }
  };

  stage_async(0, 0);
  wait_async0();
  __syncthreads();

  int cur = 0;
  for (int k0 = 0; k0 < K; k0 += 64) {
    const bool more = (k0 + 64) < K;
    if (more) stage_async(cur ^ 1, k0 + 64);   // overlaps with WMMAs below

    #pragma unroll
    for (int c = 0; c < 2; ++c) {        // two K=32 chunks in this stage
      FragBF a[4], b[2];
      #pragma unroll
      for (int i = 0; i < 4; ++i) {
        const unsigned short* base = &As[cur][(wm + i * 16 + mrow) * 64 + c * 32];
        #pragma unroll
        for (int j = 0; j < 8; ++j) {
          int kk = ((j >> 2) << 4) + kbA + ((j & 3) << 1);
          a[i].u[j] = *(const unsigned int*)&base[kk];
        }
      }
      #pragma unroll
      for (int j2 = 0; j2 < 2; ++j2) {
        const unsigned short* base = &Bs[cur][(wn + j2 * 16 + mrow) * 64 + c * 32];
        #pragma unroll
        for (int j = 0; j < 8; ++j)
          b[j2].u[j] = *(const unsigned int*)&base[((j >> 2) << 4) + kbB + ((j & 3) << 1)];
      }
      #pragma unroll
      for (int i = 0; i < 4; ++i)
        #pragma unroll
        for (int j = 0; j < 2; ++j)
          acc[i][j] = __builtin_amdgcn_wmma_f32_16x16x32_bf16(
              false, a[i].v, false, b[j].v, (short)0, acc[i][j], false, false);
    }
    if (more) {
      wait_async0();
      __syncthreads();
      cur ^= 1;
    }
  }

  // Epilogue. C fragment: VGPR v, lanes 0-15 -> M=v,N=lane; 16-31 -> M=v+8.
  // The 8 accumulator elements are M-contiguous -> packed b128 stores.
  const int nloc  = lane & 15;
  const int mbias = (lane >= 16) ? 8 : 0;
  #pragma unroll
  for (int i = 0; i < 4; ++i)
    #pragma unroll
    for (int j = 0; j < 2; ++j) {
      const int gib = m0 + wm + i * 16 + mbias;     // first of 8 M indices
      const int gt  = n0 + wn + j * 16 + nloc;      // N index (timestep)
      float val[8];
      #pragma unroll
      for (int v = 0; v < 8; ++v) val[v] = acc[i][j][v] + vec[gib + v];
      if (MODE == 0) {
        uint4 p;
        unsigned* pu = (unsigned*)&p;
        #pragma unroll
        for (int v2 = 0; v2 < 4; ++v2) {
          float lo = fmaxf(val[2 * v2], 0.f);
          float hi = fmaxf(val[2 * v2 + 1], 0.f);
          pu[v2] = (unsigned)f2bf(lo) | ((unsigned)f2bf(hi) << 16);
        }
        *(uint4*)&outBf[(size_t)gt * M + gib] = p;
      } else {
        float4 lo = {val[0], val[1], val[2], val[3]};
        float4 hi = {val[4], val[5], val[6], val[7]};
        *(float4*)&outF[(size_t)gt * M + gib]     = lo;
        *(float4*)&outF[(size_t)gt * M + gib + 4] = hi;
      }
    }
}

// ---------------------------------------------------------------------------
// K5: persistent LSTM recurrence. 32 WGs; WG b owns hidden units
//     [b*32, b*32+32) and keeps its 128x1024 W_hh slice bf16-packed in LDS
//     (256 KB of the WGP's 320 KB). Per step: async h broadcast into LDS,
//     vectorized LDS matvec, gate math, per-step arrival counter.
// ---------------------------------------------------------------------------
__global__ __launch_bounds__(256) void k_lstm(
    const float* __restrict__ Whh, const float* __restrict__ G,
    float* __restrict__ hbuf, unsigned* __restrict__ stepcnt)
{
  extern __shared__ unsigned char smem[];
  unsigned int* Wl = (unsigned int*)smem;                 // [128][512] bf16x2
  float* hs = (float*)(smem + 128 * 512 * 4);             // [1024]
  float* gd = hs + 1024;                                  // [128]
  float* cs = gd + 128;                                   // [32]

  const int tid = threadIdx.x;
  const int b   = blockIdx.x;

  // Load + convert this WG's W_hh slice: local row r = gate*32 + unit.
  for (int idx = tid; idx < 128 * 512; idx += 256) {
    int r = idx >> 9, kk = idx & 511;
    int gr = ((r >> 5) * HID) + b * 32 + (r & 31);        // global gate row
    float2 w = *(const float2*)(Whh + (size_t)gr * HID + (kk << 1));
    Wl[idx] = (unsigned)f2bf(w.x) | ((unsigned)f2bf(w.y) << 16);
  }
  if (tid < 32) cs[tid] = 0.f;
  __syncthreads();

  const int r    = tid >> 1;
  const int half = tid & 1;
  const int gr   = ((r >> 5) * HID) + b * 32 + (r & 31);  // this lane's gate row
  const uint4*  w4 = (const uint4*)&Wl[r * 512 + half * 256];
  const float4* h4 = (const float4*)&hs[half * 512];
  const unsigned hlds = lds_off(&hs[tid * 4]);            // 16B per thread

  for (int t = 0; t < SEQ; ++t) {
    const int cur = t & 1, nxt = cur ^ 1;
    async_ld_b128(hlds, &hbuf[cur * HID + tid * 4]);      // broadcast h -> LDS
    wait_async0();
    __syncthreads();

    float acc = 0.f;
    #pragma unroll 4
    for (int q = 0; q < 64; ++q) {                 // 8 bf16 MACs per b128 load
      uint4 w = w4[q];
      float4 ha = h4[2 * q], hb = h4[2 * q + 1];
      acc += bf2f(w.x) * ha.x + bf2f_hi(w.x) * ha.y
           + bf2f(w.y) * ha.z + bf2f_hi(w.y) * ha.w
           + bf2f(w.z) * hb.x + bf2f_hi(w.z) * hb.y
           + bf2f(w.w) * hb.z + bf2f_hi(w.w) * hb.w;
    }
    acc += __shfl_xor(acc, 1, 32);
    if (half == 0) {
      gd[r] = acc + G[(size_t)t * G4 + gr];
      if (t + 1 < SEQ)                              // gfx1250 global_prefetch_b8
        __builtin_prefetch(&G[(size_t)(t + 1) * G4 + gr], 0, 0);
    }
    __syncthreads();

    if (tid < 32) {                                 // PyTorch order i,f,g,o
      float ii = sigm(gd[tid]);
      float ff = sigm(gd[32 + tid]);
      float gg = tanhf(gd[64 + tid]);
      float oo = sigm(gd[96 + tid]);
      float cn = ff * cs[tid] + ii * gg;
      cs[tid] = cn;
      hbuf[nxt * HID + b * 32 + tid] = oo * tanhf(cn);
    }
    __threadfence();
    __syncthreads();
    if (tid == 0) {                                 // per-step arrival counter
      atomicAdd(&stepcnt[t], 1u);
      while (atomicAdd(&stepcnt[t], 0u) < LSTM_WGS) __builtin_amdgcn_s_sleep(1);
    }
    __syncthreads();
    __threadfence();                                // acquire other WGs' h
  }
}

// ---------------------------------------------------------------------------
// K6: out = out_W @ h_final + out_b  (7 x 1024 matvec; h_final = hbuf[0])
// ---------------------------------------------------------------------------
__global__ __launch_bounds__(256) void k_out(
    const float* __restrict__ out_W, const float* __restrict__ out_b,
    const float* __restrict__ hbuf, float* __restrict__ out)
{
  __shared__ float red[256];
  const int tid = threadIdx.x;
  for (int row = 0; row < 7; ++row) {
    float s = 0.f;
    for (int k = tid; k < HID; k += 256) s += out_W[row * HID + k] * hbuf[k];
    red[tid] = s;
    __syncthreads();
    for (int off = 128; off > 0; off >>= 1) {
      if (tid < off) red[tid] += red[tid + off];
      __syncthreads();
    }
    if (tid == 0) out[row] = red[0] + out_b[row];
    __syncthreads();
  }
}

// ---------------------------------------------------------------------------
// Host side
// ---------------------------------------------------------------------------
extern "C" void kernel_launch(void* const* d_in, const int* in_sizes, int n_in,
                              void* d_out, int out_size, void* d_ws, size_t ws_size,
                              hipStream_t stream) {
  const int*   sent_ids  = (const int*)d_in[0];
  const int*   emoji_ids = (const int*)d_in[1];
  const float* word_emb  = (const float*)d_in[2];
  const float* emoji_emb = (const float*)d_in[3];
  // d_in[4]/d_in[5] (attn_W / attn_b) unused: softmax of one logit == 1.0.
  const float* comb_W = (const float*)d_in[6];
  const float* comb_b = (const float*)d_in[7];
  const float* W_ih   = (const float*)d_in[8];
  const float* W_hh   = (const float*)d_in[9];
  const float* b_ih   = (const float*)d_in[10];
  const float* b_hh   = (const float*)d_in[11];
  const float* out_W  = (const float*)d_in[12];
  const float* out_b  = (const float*)d_in[13];
  float* out = (float*)d_out;

  char* ws = (char*)d_ws;
  // Workspace layout (bytes, 256-aligned), total ~52.5 MB.
  unsigned short* Xbf     = (unsigned short*)(ws + 0);            // 4 MB
  unsigned short* Cmbbf   = (unsigned short*)(ws + 4194304);      // 4 MB
  unsigned short* Wcxbf   = (unsigned short*)(ws + 8388608);      // 2 MB
  unsigned short* Wihbf   = (unsigned short*)(ws + 10485760);     // 8 MB
  float*          Gg      = (float*)(ws + 18874368);              // 32 MB
  float*          emojiav = (float*)(ws + 52428800);              // 4 KB
  float*          cvec    = (float*)(ws + 52432896);              // 4 KB
  float*          bsum    = (float*)(ws + 52436992);              // 16 KB
  float*          hbuf    = (float*)(ws + 52453376);              // 8 KB
  unsigned*       scnt    = (unsigned*)(ws + 52461568);           // 8 KB

  k_prep_small<<<1, 256, 0, stream>>>(emoji_ids, emoji_emb, b_ih, b_hh,
                                      emojiav, hbuf, bsum, scnt);
  k_constvec<<<4, 256, 0, stream>>>(comb_W, comb_b, emojiav, cvec);
  k_convert_gather<<<28672, 256, 0, stream>>>(sent_ids, word_emb, comb_W, W_ih,
                                              Xbf, Wcxbf, Wihbf);
  // GEMM1: combined[t][i] = relu(comb_Wx @ x_t + cvec)   M=1024,N=2048,K=1024
  k_gemm_nt<0><<<(EMB / 128) * (SEQ / 128), 256, 0, stream>>>(
      Wcxbf, Xbf, cvec, Cmbbf, nullptr, EMB, SEQ, EMB);
  // GEMM2: G[t][r] = W_ih @ combined_t + (b_ih+b_hh)     M=4096,N=2048,K=1024
  k_gemm_nt<1><<<(G4 / 128) * (SEQ / 128), 256, 0, stream>>>(
      Wihbf, Cmbbf, bsum, nullptr, Gg, G4, SEQ, EMB);
  // Persistent recurrence: 32 WGs, 256 threads, ~260.6 KB dynamic LDS each.
  const size_t lstm_lds = 128 * 512 * 4 + 1024 * 4 + 128 * 4 + 32 * 4;
  k_lstm<<<LSTM_WGS, 256, lstm_lds, stream>>>(W_hh, Gg, hbuf, scnt);
  // Final h after 2048 steps lands in parity buffer 0.
  k_out<<<1, 256, 0, stream>>>(out_W, out_b, hbuf, out);
}
